// Axial_Attention_4672924418302
// MI455X (gfx1250) — compile-verified, matrix operand order checked by reading, not run on hardware
//
#include <hip/hip_runtime.h>

// ---------------------------------------------------------------------------
// Types for CDNA5 WMMA (wave32, 16x16x32 bf16 -> f32) and TDM descriptors
// ---------------------------------------------------------------------------
typedef __attribute__((ext_vector_type(16))) __bf16 v16bf;
typedef __attribute__((ext_vector_type(8)))  float  v8f;
typedef __attribute__((ext_vector_type(4)))  unsigned int uint32x4;
typedef __attribute__((ext_vector_type(8)))  int     int32x8;
typedef __attribute__((ext_vector_type(4)))  int     int32x4;

union frag16 {
    v16bf v;
    unsigned short u[16];
};

static __device__ __forceinline__ unsigned short f32_to_bf16(float x) {
    unsigned int b = __float_as_uint(x);
    b += 0x7fffu + ((b >> 16) & 1u);          // round-to-nearest-even
    return (unsigned short)(b >> 16);
}
static __device__ __forceinline__ float bf16_to_f32(unsigned short h) {
    return __uint_as_float(((unsigned int)h) << 16);
}

static __device__ __forceinline__ v8f wmma_bf16(const frag16& a, const frag16& b, v8f c) {
    // (neg_a, A, neg_b, B, c_mod, C, reuse_a, reuse_b)
    return __builtin_amdgcn_wmma_f32_16x16x32_bf16(false, a.v, false, b.v,
                                                   (short)0, c, false, false);
}

// Per-lane A/B fragment gather for 16x32 bf16 operands (ISA 7.12.2 layout).
static __device__ __forceinline__ frag16
load_frag_row(const unsigned short* __restrict__ base, size_t stride,
              size_t row, int hi, size_t koff) {
    frag16 f;
    const unsigned short* p = base + row * stride + koff + (size_t)hi * 8;
#pragma unroll
    for (int j = 0; j < 8; ++j) { f.u[j] = p[j]; f.u[8 + j] = p[16 + j]; }
    return f;
}

// ---------------------------------------------------------------------------
// Problem constants
// ---------------------------------------------------------------------------
#define SEQ   192              // seq len for both attentions
#define BSZ   384              // batch (2*HN or 2*W)
#define CDIM  128
#define NHEAD 8
#define HD    16
#define ROWS  (SEQ * BSZ)      // 73728
#define PEROW (SEQ * SEQ)      // 36864
#define SP_LD 196              // padded logits stride (f32)

#define LDS_Q_OFF (SEQ * SP_LD * 4)            // bytes: after logits
#define LDS_K_OFF (LDS_Q_OFF + SEQ * 32 * 2)
#define LDS_V_OFF (LDS_K_OFF + SEQ * 32 * 2)

// ---------------------------------------------------------------------------
// TDM: 2-D tile (16 elems x 192 rows, 2B elements) global -> LDS.
// pad=1 inserts 32B of LDS padding after each 32B row (pad_interval=2 -> 8 DW,
// pad_amount=7 -> 8 DW), producing the 64B row stride the WMMA frag loader
// expects; pad=0 stores rows contiguously (stride 32B).
// ---------------------------------------------------------------------------
#define HAVE_TDM (__has_builtin(__builtin_amdgcn_tensor_load_to_lds) && \
                  __has_builtin(__builtin_amdgcn_s_wait_tensorcnt))

#if HAVE_TDM
static __device__ __forceinline__ void
tdm_load_tile_2d(unsigned lds_off, const void* gptr,
                 unsigned row_stride_elems, int pad) {
    const unsigned long long p = (unsigned long long)(uintptr_t)gptr;
    uint32x4 g0;
    g0[0] = 1u;                                    // count=1, user mode
    g0[1] = lds_off;                               // lds_addr (bytes)
    g0[2] = (unsigned)(p & 0xffffffffull);         // global_addr[31:0]
    g0[3] = (unsigned)((p >> 32) & 0x1ffffffull)   // global_addr[56:32]
          | (2u << 30);                            // type = 2 ("image")
    int32x8 g1;
    unsigned w0 = (1u << 16);                      // data_size = 1 -> 2 bytes
    if (pad) w0 |= (1u << 20)                      // pad_enable
               |  (2u << 22)                       // pad_interval: 8 DWORDs
               |  (7u << 25);                      // pad_amount:   8 DWORDs
    g1[0] = (int)w0;
    g1[1] = (int)(16u << 16);                      // tensor_dim0 = 16 (lo16)
    g1[2] = (int)((unsigned)SEQ << 16);            // tensor_dim1 = 192 (lo16)
    g1[3] = (int)(16u << 16);                      // tile_dim0 = 16
    g1[4] = SEQ;                                   // tile_dim1 = 192
    g1[5] = (int)row_stride_elems;                 // tensor_dim0_stride lo32
    g1[6] = 0;
    g1[7] = 0;
    const int32x4 z4 = {0, 0, 0, 0};
    const int32x8 z8 = {0, 0, 0, 0, 0, 0, 0, 0};
    __builtin_amdgcn_tensor_load_to_lds(g0, g1, z4, z4, z8, 0);
}
#endif

// ---------------------------------------------------------------------------
// 1) elementwise f32 -> bf16 convert
// ---------------------------------------------------------------------------
__global__ void cvt_bf16_kernel(const float* __restrict__ in,
                                unsigned short* __restrict__ out, int n) {
    for (int i = blockIdx.x * blockDim.x + threadIdx.x; i < n;
         i += gridDim.x * blockDim.x)
        out[i] = f32_to_bf16(in[i]);
}

// ---------------------------------------------------------------------------
// 2) LayerNorm over C + permute [W,2HN,C] -> [HN, 2W, C] (bf16 out)
// ---------------------------------------------------------------------------
__global__ __launch_bounds__(128)
void ln_permute_kernel(const float* __restrict__ feat,
                       const float* __restrict__ g, const float* __restrict__ b,
                       unsigned short* __restrict__ X2) {
    __shared__ float sred[4];
    const int m   = blockIdx.x;          // feat row
    const int c   = threadIdx.x;         // 0..127
    const int wv  = c >> 5, l = c & 31;

    const int w = m / BSZ, jj = m % BSZ, side = jj / SEQ, hn = jj % SEQ;

    float x = feat[(size_t)m * CDIM + c];

    float s = x;
#pragma unroll
    for (int o = 16; o > 0; o >>= 1) s += __shfl_xor(s, o, 32);
    if (l == 0) sred[wv] = s;
    __syncthreads();
    float mean = (sred[0] + sred[1] + sred[2] + sred[3]) * (1.0f / CDIM);
    __syncthreads();

    float d = x - mean;
    float v = d * d;
#pragma unroll
    for (int o = 16; o > 0; o >>= 1) v += __shfl_xor(v, o, 32);
    if (l == 0) sred[wv] = v;
    __syncthreads();
    float var = (sred[0] + sred[1] + sred[2] + sred[3]) * (1.0f / CDIM);

    float y = d * __frsqrt_rn(var + 1e-5f) * g[c] + b[c];

    const size_t dst = ((size_t)hn * BSZ + (size_t)side * SEQ + w) * CDIM + c;
    X2[dst] = f32_to_bf16(y);
}

// ---------------------------------------------------------------------------
// 3) Generic WMMA GEMM:  C[M,N] = A[M,K] @ B[N,K]^T + bias
//    block = 256 threads = 8 waves, tile 128(M) x 64(N), K in steps of 32.
//    B fragments batched per K-step -> one wait, 4 back-to-back WMMAs.
//    mode 0: bf16 store; mode 1: bf16 + axial re-permute; mode 2: f32 + resid
// ---------------------------------------------------------------------------
__global__ __launch_bounds__(256)
void gemm_wmma_kernel(const unsigned short* __restrict__ A,
                      const unsigned short* __restrict__ B,
                      const float* __restrict__ bias,
                      const int* __restrict__ aidx,
                      void* __restrict__ Cout,
                      const float* __restrict__ resid,
                      int M, int N, int K, int mode) {
    const int wv = threadIdx.x >> 5;
    const int l  = threadIdx.x & 31;
    const int hi = l >> 4, ln = l & 15;

    const int mBase = blockIdx.x * 128 + wv * 16;
    const int nBase = blockIdx.y * 64;

    const v8f zero = {0.f, 0.f, 0.f, 0.f, 0.f, 0.f, 0.f, 0.f};
    v8f acc[4];
#pragma unroll
    for (int t = 0; t < 4; ++t) acc[t] = zero;

    const int arow = mBase + ln;
    const size_t asrc = aidx ? (size_t)aidx[arow] : (size_t)arow;

    for (int k0 = 0; k0 < K; k0 += 32) {
        frag16 af = load_frag_row(A, (size_t)K, asrc, hi, (size_t)k0);
        frag16 bfr[4];
#pragma unroll
        for (int t = 0; t < 4; ++t)
            bfr[t] = load_frag_row(B, (size_t)K, (size_t)(nBase + t * 16 + ln),
                                   hi, (size_t)k0);
#pragma unroll
        for (int t = 0; t < 4; ++t)
            acc[t] = wmma_bf16(af, bfr[t], acc[t]);
    }

#pragma unroll
    for (int t = 0; t < 4; ++t) {
        const int col = nBase + t * 16 + ln;
        const float bb = bias ? bias[col] : 0.0f;
#pragma unroll
        for (int r = 0; r < 8; ++r) {
            const int row = mBase + hi * 8 + r;
            const float val = acc[t][r] + bb;
            if (mode == 0) {
                ((unsigned short*)Cout)[(size_t)row * N + col] = f32_to_bf16(val);
            } else if (mode == 1) {
                // row = hn*384 + side*192 + w  ->  w*384 + side*192 + hn
                const int a = row / BSZ, rem = row % BSZ;
                const int side = rem / SEQ, tt = rem % SEQ;
                const size_t dst = ((size_t)tt * BSZ + (size_t)side * SEQ + a) * CDIM + col;
                ((unsigned short*)Cout)[dst] = f32_to_bf16(val);
            } else {
                const size_t dst = (size_t)row * CDIM + col;
                ((float*)Cout)[dst] = resid[dst] + val;
            }
        }
    }
}

// ---------------------------------------------------------------------------
// 4) Fused rel-pos attention core, one workgroup per (batch n, head e).
//    LDS: logits f32 [192][196] + Q,K bf16 [192][32] (K padded w/ zeros)
//         + V bf16 [192][16]  => 181,248 B (one WG per 320KB WGP).
//    Q/K/V staged by the Tensor Data Mover (pad_enable builds the 64B row
//    stride); hd^-0.5 folded into the logits pass so staging is a pure copy.
// ---------------------------------------------------------------------------
__global__ __launch_bounds__(256)
void attn_core_kernel(const unsigned short* __restrict__ QKV,  // [ROWS][384]
                      const unsigned short* __restrict__ PE,   // [PEROW][256]
                      unsigned short* __restrict__ AO) {       // [ROWS][128]
    extern __shared__ char smem_raw[];
    float*          sP = (float*)smem_raw;                       // 192*196
    unsigned short* sQ = (unsigned short*)(smem_raw + LDS_Q_OFF);
    unsigned short* sK = (unsigned short*)(smem_raw + LDS_K_OFF);
    unsigned short* sV = (unsigned short*)(smem_raw + LDS_V_OFF);

    const int n = blockIdx.x >> 3;
    const int e = blockIdx.x & 7;
    const int tid = threadIdx.x;
    const int wv = tid >> 5, l = tid & 31;
    const int hi = l >> 4, ln = l & 15;

    // ---- zero the K-pad columns (TDM skips them; disjoint from TDM writes)
    for (int i = tid; i < SEQ; i += 256) {
#pragma unroll
        for (int j = 0; j < 8; ++j) {
            ((unsigned int*)(sQ + i * 32 + 16))[j] = 0u;
            ((unsigned int*)(sK + i * 32 + 16))[j] = 0u;
        }
    }

#if HAVE_TDM
    if (wv == 0) {   // one wave drives the TDM; EXEC is ignored by tensor ops
        const unsigned short* qb = QKV + (size_t)n * (3 * CDIM) + e * HD;
        tdm_load_tile_2d(LDS_Q_OFF, qb,            BSZ * 3 * CDIM, 1);
        tdm_load_tile_2d(LDS_K_OFF, qb + CDIM,     BSZ * 3 * CDIM, 1);
        tdm_load_tile_2d(LDS_V_OFF, qb + 2 * CDIM, BSZ * 3 * CDIM, 0);
        __builtin_amdgcn_s_wait_tensorcnt(0);
    }
#else
    for (int i = tid; i < SEQ * HD; i += 256) {
        const int s = i >> 4, c = i & 15;
        const unsigned short* base =
            QKV + ((size_t)s * BSZ + n) * (3 * CDIM) + e * HD + c;
        sQ[s * 32 + c] = base[0];
        sK[s * 32 + c] = base[CDIM];
        sV[s * 16 + c] = base[2 * CDIM];
    }
#endif
    __syncthreads();

    // ---- term1: logits = Q @ K^T (unscaled; 144 16x16 tiles over 8 waves)
    for (int t = wv; t < 144; t += 8) {
        const int wi = t / 12, vi = t % 12;
        frag16 af  = load_frag_row(sQ, 32, (size_t)(wi * 16 + ln), hi, 0);
        frag16 bfr = load_frag_row(sK, 32, (size_t)(vi * 16 + ln), hi, 0);
        v8f d = {0.f, 0.f, 0.f, 0.f, 0.f, 0.f, 0.f, 0.f};
        d = wmma_bf16(af, bfr, d);
        const int col = vi * 16 + ln;
        const int rb = wi * 16 + hi * 8;
#pragma unroll
        for (int r = 0; r < 8; ++r) sP[(rb + r) * SP_LD + col] = d[r];
    }
    __syncthreads();

    // ---- terms 2+3 and global hd^-0.5 scale:
    //      logits = 0.25 * (qk + q·k_r + k·q_r)     (PE is L2-resident)
    for (int i = tid; i < PEROW; i += 256) {
        const int w = i / SEQ, v = i % SEQ;
        const unsigned short* pe = PE + (size_t)i * 256 + e * HD;
        __builtin_prefetch((const void*)(pe + 256 * 256), 0, 0);
        float a2 = 0.f, a3 = 0.f;
#pragma unroll
        for (int c = 0; c < HD; ++c) {
            a2 += bf16_to_f32(sQ[w * 32 + c]) * bf16_to_f32(pe[CDIM + c]); // q·k_r
            a3 += bf16_to_f32(sK[v * 32 + c]) * bf16_to_f32(pe[c]);        // k·q_r
        }
        sP[w * SP_LD + v] = 0.25f * (sP[w * SP_LD + v] + a2 + a3);
    }
    __syncthreads();

    // ---- softmax over v (one wave per row, 6 elems/lane)
    for (int w = wv; w < SEQ; w += 8) {
        float* row = sP + w * SP_LD;
        float vals[6];
        float mx = -1e30f;
#pragma unroll
        for (int j = 0; j < 6; ++j) { vals[j] = row[l + j * 32]; mx = fmaxf(mx, vals[j]); }
#pragma unroll
        for (int o = 16; o > 0; o >>= 1) mx = fmaxf(mx, __shfl_xor(mx, o, 32));
        float ss = 0.f;
#pragma unroll
        for (int j = 0; j < 6; ++j) { vals[j] = __expf(vals[j] - mx); ss += vals[j]; }
#pragma unroll
        for (int o = 16; o > 0; o >>= 1) ss += __shfl_xor(ss, o, 32);
        const float inv = 1.0f / ss;
#pragma unroll
        for (int j = 0; j < 6; ++j) row[l + j * 32] = vals[j] * inv;
    }
    __syncthreads();

    // ---- out = P @ V   (12 tiles of 16(w) x 16(c), K = 192 over v)
    for (int wi = wv; wi < 12; wi += 8) {
        v8f acc = {0.f, 0.f, 0.f, 0.f, 0.f, 0.f, 0.f, 0.f};
        for (int kk = 0; kk < 6; ++kk) {
            const int vb = kk * 32 + hi * 8;
            frag16 af, bfr;
            const int w = wi * 16 + ln;
#pragma unroll
            for (int j = 0; j < 8; ++j) {
                af.u[j]      = f32_to_bf16(sP[w * SP_LD + vb + j]);
                af.u[8 + j]  = f32_to_bf16(sP[w * SP_LD + vb + 16 + j]);
                bfr.u[j]     = sV[(vb + j) * 16 + ln];
                bfr.u[8 + j] = sV[(vb + 16 + j) * 16 + ln];
            }
            acc = wmma_bf16(af, bfr, acc);
        }
#pragma unroll
        for (int r = 0; r < 8; ++r) {
            const int s = wi * 16 + hi * 8 + r;
            AO[((size_t)s * BSZ + n) * CDIM + e * HD + ln] = f32_to_bf16(acc[r]);
        }
    }
}

// ---------------------------------------------------------------------------
// host launcher
// ---------------------------------------------------------------------------
extern "C" void kernel_launch(void* const* d_in, const int* in_sizes, int n_in,
                              void* d_out, int out_size, void* d_ws, size_t ws_size,
                              hipStream_t stream) {
    (void)n_in; (void)out_size; (void)ws_size;
    const float* feat   = (const float*)d_in[0];
    const float* pos    = (const float*)d_in[1];
    const float* pos_y  = (const float*)d_in[2];
    const float* ln_w   = (const float*)d_in[3];
    const float* ln_b   = (const float*)d_in[4];
    const float* w_in1  = (const float*)d_in[5];
    const float* b_in1  = (const float*)d_in[6];
    const float* w_out1 = (const float*)d_in[7];
    const float* b_out1 = (const float*)d_in[8];
    const float* w_in2  = (const float*)d_in[9];
    const float* b_in2  = (const float*)d_in[10];
    const float* w_out2 = (const float*)d_in[11];
    const float* b_out2 = (const float*)d_in[12];
    const int*   idx_x  = (const int*)d_in[13];
    const int*   idx_y  = (const int*)d_in[14];

    // workspace carve-up (bytes)
    char* ws = (char*)d_ws;
    size_t off = 0;
    auto take = [&](size_t bytes) { char* p = ws + off; off += (bytes + 255) & ~(size_t)255; return p; };
    unsigned short* Xbuf   = (unsigned short*)take((size_t)ROWS * CDIM * 2);      // 18.9 MB
    unsigned short* QKV    = (unsigned short*)take((size_t)ROWS * 3 * CDIM * 2);  // 56.6 MB
    unsigned short* PEbuf  = (unsigned short*)take((size_t)PEROW * 256 * 2);      // 18.9 MB
    unsigned short* AO     = (unsigned short*)take((size_t)ROWS * CDIM * 2);      // 18.9 MB
    unsigned short* bw_in1 = (unsigned short*)take((size_t)in_sizes[5] * 2);
    unsigned short* bw_in2 = (unsigned short*)take((size_t)in_sizes[9] * 2);
    unsigned short* bw_o1  = (unsigned short*)take((size_t)in_sizes[7] * 2);
    unsigned short* bw_o2  = (unsigned short*)take((size_t)in_sizes[11] * 2);
    unsigned short* bpos   = (unsigned short*)take((size_t)in_sizes[1] * 2);
    unsigned short* bpos_y = (unsigned short*)take((size_t)in_sizes[2] * 2);

    // 1) convert weights / pos tables to bf16
    auto cvt = [&](const float* s, unsigned short* d, int n) {
        cvt_bf16_kernel<<<(n + 255) / 256, 256, 0, stream>>>(s, d, n);
    };
    cvt(w_in1,  bw_in1, in_sizes[5]);
    cvt(w_in2,  bw_in2, in_sizes[9]);
    cvt(w_out1, bw_o1,  in_sizes[7]);
    cvt(w_out2, bw_o2,  in_sizes[11]);
    cvt(pos,    bpos,   in_sizes[1]);
    cvt(pos_y,  bpos_y, in_sizes[2]);

    // 2) LN + vertical permute -> X2
    ln_permute_kernel<<<ROWS, 128, 0, stream>>>(feat, ln_w, ln_b, Xbuf);

    const size_t attn_lds = (size_t)LDS_V_OFF + (size_t)SEQ * 16 * 2; // 181,248 B

    // ---------------- attention 2 (vertical, uses pos_y / w_in2) ----------
    gemm_wmma_kernel<<<dim3(ROWS / 128, 384 / 64), 256, 0, stream>>>(
        Xbuf, bw_in2, b_in2, nullptr, QKV, nullptr, ROWS, 384, CDIM, 0);
    gemm_wmma_kernel<<<dim3(PEROW / 128, 256 / 64), 256, 0, stream>>>(
        bpos_y, bw_in2, b_in2, idx_y, PEbuf, nullptr, PEROW, 256, CDIM, 0);
    attn_core_kernel<<<BSZ * NHEAD, 256, attn_lds, stream>>>(QKV, PEbuf, AO);
    gemm_wmma_kernel<<<dim3(ROWS / 128, 128 / 64), 256, 0, stream>>>(
        AO, bw_o2, b_out2, nullptr, Xbuf, nullptr, ROWS, CDIM, CDIM, 1); // permute -> X1

    // ---------------- attention 1 (horizontal, uses pos / w_in1) ----------
    gemm_wmma_kernel<<<dim3(ROWS / 128, 384 / 64), 256, 0, stream>>>(
        Xbuf, bw_in1, b_in1, nullptr, QKV, nullptr, ROWS, 384, CDIM, 0);
    gemm_wmma_kernel<<<dim3(PEROW / 128, 256 / 64), 256, 0, stream>>>(
        bpos, bw_in1, b_in1, idx_x, PEbuf, nullptr, PEROW, 256, CDIM, 0);
    attn_core_kernel<<<BSZ * NHEAD, 256, attn_lds, stream>>>(QKV, PEbuf, AO);
    gemm_wmma_kernel<<<dim3(ROWS / 128, 128 / 64), 256, 0, stream>>>(
        AO, bw_o1, b_out1, nullptr, d_out, feat, ROWS, CDIM, CDIM, 2); // + residual
}
